// RayUpsampler_4380866642610
// MI455X (gfx1250) — matrix operand order repaired and checked by reading, not compiled, not run on hardware
//
#include <hip/hip_runtime.h>

// ---------------------------------------------------------------------------
// FIR-upsample(x2) + 3x3 conv, fused pipeline for MI455X (gfx1250, wave32).
//   155 GFLOP conv -> compute bound on v_wmma_f32_16x16x32_f16 (~30us).
//   f16 intermediate ypad (~72MB) fits in 192MB L2.
//   A-tile staging uses GLOBAL_LOAD_ASYNC_TO_LDS_B128 (ASYNCcnt), B-tile
//   transpose uses conflict-free b32 LDS scatters.
// ---------------------------------------------------------------------------

typedef __attribute__((ext_vector_type(16))) _Float16 v16h;
typedef __attribute__((ext_vector_type(8)))  _Float16 v8h;
typedef __attribute__((ext_vector_type(2)))  _Float16 h2;
typedef __attribute__((ext_vector_type(8)))  float    v8f;

#define NN    8
#define CCH   256     // channels in/out
#define HIN   64
#define HOUT  128     // output H = W
#define PADH  130     // ypad rows (1 halo row each side)
#define PADW  136     // ypad cols (272B row stride, 16B aligned)
#define CHS   (PADH * PADW)   // per-channel ypad elements
#define LSTR  40      // LDS row stride in f16 (80B: 16B-aligned, bank-safe)

#define WH_ELEMS  (9 * CCH * CCH)
#define YPAD_ELEMS ((size_t)NN * CCH * PADH * PADW)

// ---------------------------------------------------------------------------
// Kernel 1: repack fp32 conv_w [co][ci][3][3] -> f16 wh [tap][co][ci]
// ---------------------------------------------------------------------------
__global__ __launch_bounds__(256) void ru_prep_w(const float* __restrict__ w,
                                                 _Float16* __restrict__ wh) {
    int idx = blockIdx.x * 256 + threadIdx.x;
    int co = idx >> 8, ci = idx & 255;
#pragma unroll
    for (int r = 0; r < 3; ++r)
#pragma unroll
        for (int s = 0; s < 3; ++s)
            wh[((size_t)((r * 3 + s) * CCH + co)) * CCH + ci] =
                (_Float16)w[((size_t)(co * CCH + ci) * 3 + r) * 3 + s];
}

// ---------------------------------------------------------------------------
// Kernel 2: fused zero-insert upsample + [1,3,3,1] FIR (separable), fp32->f16.
//   y[2i] = (x[i-1]+3x[i])/4 ; y[2i+1] = (3x[i]+x[i+1])/4 (per axis)
// ---------------------------------------------------------------------------
__global__ __launch_bounds__(256) void ru_upsample(const float* __restrict__ x,
                                                   _Float16* __restrict__ ypad) {
    unsigned idx = blockIdx.x * 256u + threadIdx.x;
    int q = idx & 127, p = (idx >> 7) & 127;
    int c = (idx >> 14) & 255, n = idx >> 22;
    const float* xp = x + ((size_t)(n * CCH + c)) * (HIN * HIN);

    int i = p >> 1, j = q >> 1;
    int r0, r1, c0, c1; float wr0, wr1, wc0, wc1;
    if (p & 1) { r0 = i;     r1 = i + 1; wr0 = 0.75f; wr1 = 0.25f; }
    else       { r0 = i - 1; r1 = i;     wr0 = 0.25f; wr1 = 0.75f; }
    if (q & 1) { c0 = j;     c1 = j + 1; wc0 = 0.75f; wc1 = 0.25f; }
    else       { c0 = j - 1; c1 = j;     wc0 = 0.25f; wc1 = 0.75f; }

    float v00 = (r0 >= 0 && c0 >= 0)  ? xp[r0 * HIN + c0] : 0.f;
    float v01 = (r0 >= 0 && c1 < HIN) ? xp[r0 * HIN + c1] : 0.f;
    float v10 = (r1 < HIN && c0 >= 0) ? xp[r1 * HIN + c0] : 0.f;
    float v11 = (r1 < HIN && c1 < HIN)? xp[r1 * HIN + c1] : 0.f;

    float v = wr0 * (wc0 * v00 + wc1 * v01) + wr1 * (wc0 * v10 + wc1 * v11);
    ypad[((size_t)(n * CCH + c) * PADH + (p + 1)) * PADW + (q + 1)] = (_Float16)v;
}

// ---------------------------------------------------------------------------
// Kernel 3: implicit-GEMM 3x3 conv with v_wmma_f32_16x16x32_f16.
//   Block: 128 cout x 128 spatial (one output row), 8 waves in 2x4.
//   Per (r,kc): A tiles (3 taps) via async global->LDS; one wide transposed
//   B tile Bt[q'=0..129][k=0..31] via conflict-free b32 scatters.
// ---------------------------------------------------------------------------
__global__ __launch_bounds__(256) void ru_conv_wmma(const _Float16* __restrict__ wh,
                                                    const _Float16* __restrict__ ypad,
                                                    const float* __restrict__ bias,
                                                    float* __restrict__ out) {
    const int tid  = threadIdx.x;
    const int lane = tid & 31;
    const int wave = tid >> 5;
    const int l15  = lane & 15, hi = lane >> 4;
    const int wm   = wave >> 2;          // 0..1 (64 cout rows each)
    const int wn   = wave & 3;           // 0..3 (32 q cols each)

    const int m0 = blockIdx.x * 128;
    const int np = blockIdx.y;
    const int n  = np >> 7, p = np & 127;

    __shared__ _Float16 lds[3 * 128 * LSTR + PADH * LSTR];
    _Float16* As = lds;                      // [s][128][LSTR]
    _Float16* Bt = lds + 3 * 128 * LSTR;     // [q'][LSTR]

    v8f acc[4][2];
#pragma unroll
    for (int im = 0; im < 4; ++im)
#pragma unroll
        for (int in = 0; in < 2; ++in) acc[im][in] = (v8f){};

    // B staging: half-wave-uniform q'-chunk, lanes span channel pairs -> b32 scatters
    const int c2 = (tid & 15) * 2;       // channel pair base 0,2,..,30
    const int jb = tid >> 4;             // 0..15, covers q' = jb*8 .. jb*8+7
    // A staging: per-thread byte offset inside the (tap,kc) weight tile
    const int arow = tid >> 1, ahalf = tid & 1;
    const unsigned avoff = (unsigned)(arow * 512 + ahalf * 32);   // 256 f16 rows

    for (int r = 0; r < 3; ++r) {
        const size_t brow = ((size_t)(n * CCH) * PADH + (size_t)(p + r)) * PADW;
        for (int kc = 0; kc < CCH; kc += 32) {
            // ---- B global loads for this step (overlap prev step's WMMAs) ----
            const _Float16* bsrc = ypad + brow + (size_t)(kc + c2) * CHS + jb * 8;
            v8h b0 = *(const v8h*)bsrc;              // channel c2,   8 cols
            v8h b1 = *(const v8h*)(bsrc + CHS);      // channel c2+1, 8 cols
            _Float16 t00 = bsrc[8], t01 = bsrc[9];   // tail cols (used by jb==15)
            _Float16 t10 = bsrc[CHS + 8], t11 = bsrc[CHS + 9];
            if (kc < CCH - 32)
                __builtin_prefetch(bsrc + 32 * CHS, 0, 1);

            __syncthreads();                          // prev compute done

            // ---- A tiles: async global -> LDS (ASYNCcnt), 3 taps ----
#pragma unroll
            for (int s = 0; s < 3; ++s) {
                unsigned long long abase = (unsigned long long)(const void*)
                    (wh + ((size_t)((r * 3 + s) * CCH + m0)) * CCH + kc);
                unsigned aldst = (unsigned)(uintptr_t)(const void*)
                    (As + s * 128 * LSTR + arow * LSTR + ahalf * 16);
                asm volatile("global_load_async_to_lds_b128 %0, %1, %2 offset:0"
                             :: "v"(aldst), "v"(avoff), "s"(abase) : "memory");
                asm volatile("global_load_async_to_lds_b128 %0, %1, %2 offset:16"
                             :: "v"(aldst), "v"(avoff), "s"(abase) : "memory");
            }

            // ---- B transpose scatter: packed 2ch b32, bank-conflict free ----
#pragma unroll
            for (int e = 0; e < 8; ++e) {
                h2 hv; hv[0] = b0[e]; hv[1] = b1[e];
                *(h2*)&Bt[(jb * 8 + e) * LSTR + c2] = hv;
            }
            if (jb == 15) {                           // q' = 128, 129 tail
                h2 ha; ha[0] = t00; ha[1] = t10;
                h2 hb; hb[0] = t01; hb[1] = t11;
                *(h2*)&Bt[128 * LSTR + c2] = ha;
                *(h2*)&Bt[129 * LSTR + c2] = hb;
            }

            asm volatile("s_wait_asynccnt 0x0" ::: "memory");
            __syncthreads();                          // staging visible

            // ---- 3 taps x (4x2) WMMA fragments ----
#pragma unroll
            for (int s = 0; s < 3; ++s) {
                v16h af[4], bf[2];
#pragma unroll
                for (int im = 0; im < 4; ++im) {
                    // A 16x32: lo lanes K{0..7,16..23}, hi lanes K{8..15,24..31}
                    const _Float16* pA = As + s * 128 * LSTR
                                       + (wm * 64 + im * 16 + l15) * LSTR + hi * 8;
                    v8h lo = *(const v8h*)pA;
                    v8h hh = *(const v8h*)(pA + 16);
                    af[im] = __builtin_shufflevector(lo, hh, 0, 1, 2, 3, 4, 5, 6, 7,
                                                     8, 9, 10, 11, 12, 13, 14, 15);
                }
#pragma unroll
                for (int in = 0; in < 2; ++in) {
                    // B 32x16: lo lanes K{0..7,16..23}, hi lanes K{8..15,24..31}
                    const _Float16* pB = Bt + (wn * 32 + in * 16 + l15 + s) * LSTR
                                       + hi * 8;
                    v8h lo = *(const v8h*)pB;
                    v8h hh = *(const v8h*)(pB + 16);
                    bf[in] = __builtin_shufflevector(lo, hh, 0, 1, 2, 3, 4, 5, 6, 7,
                                                     8, 9, 10, 11, 12, 13, 14, 15);
                }
#pragma unroll
                for (int im = 0; im < 4; ++im)
#pragma unroll
                    for (int in = 0; in < 2; ++in)
                        acc[im][in] = __builtin_amdgcn_wmma_f32_16x16x32_f16(
                            false, af[im], false, bf[in],
                            (short)0, acc[im][in], false, false);
            }
        }
    }

    // ---- epilogue: D layout VGPR v -> M = v + hi*8, N = l15 ----
#pragma unroll
    for (int im = 0; im < 4; ++im)
#pragma unroll
        for (int in = 0; in < 2; ++in) {
            const int q = wn * 32 + in * 16 + l15;
#pragma unroll
            for (int v = 0; v < 8; ++v) {
                const int co = m0 + wm * 64 + im * 16 + hi * 8 + v;
                out[(((size_t)n * CCH + co) * HOUT + p) * HOUT + q] =
                    acc[im][in][v] + bias[co];
            }
        }
}

// ---------------------------------------------------------------------------
extern "C" void kernel_launch(void* const* d_in, const int* in_sizes, int n_in,
                              void* d_out, int out_size, void* d_ws, size_t ws_size,
                              hipStream_t stream) {
    const float* x      = (const float*)d_in[0];   // [8,256,64,64]
    const float* conv_w = (const float*)d_in[1];   // [256,256,3,3]
    const float* conv_b = (const float*)d_in[2];   // [256]

    char* ws = (char*)d_ws;
    _Float16* wh   = (_Float16*)ws;                          // 1.18 MB
    _Float16* ypad = (_Float16*)(ws + (size_t)WH_ELEMS * 2); // ~72.4 MB

    hipMemsetAsync(ypad, 0, YPAD_ELEMS * sizeof(_Float16), stream);

    ru_prep_w<<<(CCH * CCH) / 256, 256, 0, stream>>>(conv_w, wh);

    const unsigned ytotal = NN * CCH * HOUT * HOUT;
    ru_upsample<<<ytotal / 256, 256, 0, stream>>>(x, ypad);

    ru_conv_wmma<<<dim3(CCH / 128, NN * HOUT), 256, 0, stream>>>(
        wh, ypad, conv_b, (float*)d_out);
}
